// TransformerBlock_1975684956816
// MI455X (gfx1250) — compile-verified, matrix operand order checked by reading, not run on hardware
//
#include <hip/hip_runtime.h>
#include <hip/hip_bf16.h>
#include <math.h>

// ---------------- problem constants ----------------
#define D_MODEL 1024
#define NH      16
#define HD      64
#define SEQ     2048
#define BATCH   2
#define ROWS    (BATCH*SEQ)          // 4096

typedef __attribute__((ext_vector_type(16))) _Float16 v16h;
typedef __attribute__((ext_vector_type(8)))  _Float16 v8h;
typedef __attribute__((ext_vector_type(8)))  float    v8f;

// Load a 16-bit A/B WMMA fragment from a row-major LDS (or global) row.
// Per CDNA5 ISA 16-bit 16x32 layout: lane L holds row/col = L%16,
// K = khalf..khalf+7 and 16+khalf..16+khalf+7, khalf = (L>=16)?8:0.
__device__ __forceinline__ v16h frag_ld(const _Float16* p, int khalf) {
  v8h lo = *(const v8h*)(p + khalf);
  v8h hi = *(const v8h*)(p + khalf + 16);
  v16h r;
#pragma unroll
  for (int i = 0; i < 8; i++) { r[i] = lo[i]; r[i + 8] = hi[i]; }
  return r;
}

// ---- DPP16 16-lane butterfly reductions (stay within one DPP row: lanes
// 0-15 / 16-31 reduce independently, matching the WMMA C-layout halves). ----
template <int CTRL>
__device__ __forceinline__ float dpp_mov(float x) {
  return __int_as_float(__builtin_amdgcn_update_dpp(
      0, __float_as_int(x), CTRL, 0xf, 0xf, true));
}
__device__ __forceinline__ float red16_max(float x) {
  x = fmaxf(x, dpp_mov<0xB1>(x));   // quad_perm [1,0,3,2]  (xor 1)
  x = fmaxf(x, dpp_mov<0x4E>(x));   // quad_perm [2,3,0,1]  (xor 2)
  x = fmaxf(x, dpp_mov<0x141>(x));  // row_half_mirror      (combine quads)
  x = fmaxf(x, dpp_mov<0x140>(x));  // row_mirror           (combine octs)
  return x;
}
__device__ __forceinline__ float red16_sum(float x) {
  x += dpp_mov<0xB1>(x);
  x += dpp_mov<0x4E>(x);
  x += dpp_mov<0x141>(x);
  x += dpp_mov<0x140>(x);
  return x;
}

// Async global->LDS staging (CDNA5 ASYNCcnt path): per-lane 16B transfer,
// no VGPR transit. ldsOff = addrspace(3) byte offset, gp = per-lane address.
#define ASYNC_LD_B128(ldsOff, gp)                                            \
  asm volatile("global_load_async_to_lds_b128 %0, %1, off"                   \
               :: "v"(ldsOff), "v"(gp) : "memory")
#define WAIT_ASYNC() asm volatile("s_wait_asynccnt 0x0" ::: "memory")

__device__ __forceinline__ unsigned lds_off(const void* p) {
  return (unsigned)(size_t)p;  // low 32 bits of flat LDS address = LDS offset
}

// ---------------- f32 -> f16 conversion ----------------
__global__ void f32_to_f16_kernel(const float* __restrict__ in,
                                  _Float16* __restrict__ out, int n) {
  int i = blockIdx.x * blockDim.x + threadIdx.x;
  if (i < n) out[i] = (_Float16)in[i];
}

// ---------------- RMSNorm: f32 row -> f16 row ----------------
__global__ __launch_bounds__(256)
void rmsnorm_kernel(const float* __restrict__ X, const float* __restrict__ g,
                    _Float16* __restrict__ Y) {
  __shared__ float red[8];
  const int row = blockIdx.x;
  const float* xr = X + (size_t)row * D_MODEL;
  float ss = 0.f;
  for (int c = threadIdx.x; c < D_MODEL; c += 256) { float v = xr[c]; ss += v * v; }
#pragma unroll
  for (int off = 16; off; off >>= 1) ss += __shfl_xor(ss, off, 32);
  const int lane = threadIdx.x & 31, wv = threadIdx.x >> 5;
  if (lane == 0) red[wv] = ss;
  __syncthreads();
  if (wv == 0) {
    float t = (lane < 8) ? red[lane] : 0.f;
#pragma unroll
    for (int off = 4; off; off >>= 1) t += __shfl_xor(t, off, 32);
    if (lane == 0) red[0] = t;
  }
  __syncthreads();
  const float inv = rsqrtf(red[0] / (float)D_MODEL + 1e-6f);
  for (int c = threadIdx.x; c < D_MODEL; c += 256)
    Y[(size_t)row * D_MODEL + c] = (_Float16)(xr[c] * inv * g[c]);
}

// ---------------- generic WMMA GEMM: C[M,N] = A[M,K]*B[K,N] + bias ----------------
// Double-buffered pipeline: async A(k+1)->LDS + global B(k+1)->regs are in
// flight while WMMAs run on buffer k; B transpose ds_stores land after the
// math; one s_wait_asynccnt + barrier per K tile.
// EPI 0: store f32   EPI 1: store f32 + residual   EPI 2: exact GELU -> f16
#define BM 128
#define BN 64
#define BK 32
#define ASTR 40   // halves, 16B-aligned row stride (pad vs bank conflicts)
#define BSTR 40

template <int EPI>
__global__ __launch_bounds__(256)
void gemm_wmma(const _Float16* __restrict__ A, const _Float16* __restrict__ B,
               const float* __restrict__ bias, const float* __restrict__ resid,
               float* __restrict__ Cf, _Float16* __restrict__ Ch,
               int M, int N, int K) {
  __shared__ __align__(16) _Float16 As[2][BM * ASTR];
  __shared__ __align__(16) _Float16 Bs[2][BN * BSTR];  // transposed: Bs[n][k]
  const int tid   = threadIdx.x;
  const int lane  = tid & 31;
  const int wave  = tid >> 5;
  const int wm    = wave & 3;          // wave row -> m offset wm*32
  const int wn    = wave >> 2;         // wave col -> n offset wn*32
  const int m0    = blockIdx.y * BM;
  const int n0    = blockIdx.x * BN;
  const int lr    = lane & 15;
  const int khalf = (lane >> 4) << 3;  // 0 or 8

  // staging geometry
  const int arow = tid >> 2;                 // A chunk row (two chunks/thread)
  const int acol = (tid & 3) << 3;           // A chunk col (halves)
  const int bkr  = tid >> 3;                 // B k-row
  const int bnc  = (tid & 7) << 3;           // B n-col (halves)

  v8f acc[2][2];
#pragma unroll
  for (int i = 0; i < 2; i++)
#pragma unroll
    for (int j = 0; j < 2; j++) acc[i][j] = (v8f)0.f;

  // ---- prologue: stage tile 0 into buffer 0 ----
  {
    ASYNC_LD_B128(lds_off(&As[0][arow * ASTR + acol]),
                  A + (size_t)(m0 + arow) * K + acol);
    ASYNC_LD_B128(lds_off(&As[0][(arow + 64) * ASTR + acol]),
                  A + (size_t)(m0 + arow + 64) * K + acol);
    v8h breg = *(const v8h*)(B + (size_t)bkr * N + n0 + bnc);
#pragma unroll
    for (int j = 0; j < 8; j++) Bs[0][(bnc + j) * BSTR + bkr] = breg[j];
  }
  WAIT_ASYNC();
  __syncthreads();

  const int ntiles = K / BK;
  for (int kt = 0; kt < ntiles; kt++) {
    const int  cur  = kt & 1;
    const bool more = (kt + 1 < ntiles);
    v8h breg;
    if (more) {  // issue next-tile loads before doing the math
      const int k1 = (kt + 1) * BK;
      ASYNC_LD_B128(lds_off(&As[cur ^ 1][arow * ASTR + acol]),
                    A + (size_t)(m0 + arow) * K + k1 + acol);
      ASYNC_LD_B128(lds_off(&As[cur ^ 1][(arow + 64) * ASTR + acol]),
                    A + (size_t)(m0 + arow + 64) * K + k1 + acol);
      breg = *(const v8h*)(B + (size_t)(k1 + bkr) * N + n0 + bnc);
    }

    v16h af[2], bf[2];
#pragma unroll
    for (int i = 0; i < 2; i++)
      af[i] = frag_ld(&As[cur][(wm * 32 + i * 16 + lr) * ASTR], khalf);
#pragma unroll
    for (int j = 0; j < 2; j++)
      bf[j] = frag_ld(&Bs[cur][(wn * 32 + j * 16 + lr) * BSTR], khalf);
#pragma unroll
    for (int i = 0; i < 2; i++)
#pragma unroll
      for (int j = 0; j < 2; j++)
        acc[i][j] = __builtin_amdgcn_wmma_f32_16x16x32_f16(
            false, af[i], false, bf[j], (short)0, acc[i][j], false, false);

    if (more) {  // transpose-store B after the math hides its load latency
#pragma unroll
      for (int j = 0; j < 8; j++) Bs[cur ^ 1][(bnc + j) * BSTR + bkr] = breg[j];
      WAIT_ASYNC();
    }
    __syncthreads();
  }

  // epilogue: acc VGPR r holds M = r + 8*(lane>=16), N = lane%16
  const int mrow_base = m0 + wm * 32 + khalf;
  const int ncol_base = n0 + wn * 32 + lr;
#pragma unroll
  for (int i = 0; i < 2; i++) {
#pragma unroll
    for (int j = 0; j < 2; j++) {
      const int   nc = ncol_base + j * 16;
      const float bv = bias[nc];
#pragma unroll
      for (int r = 0; r < 8; r++) {
        const int    mr  = mrow_base + i * 16 + r;
        const size_t idx = (size_t)mr * N + nc;
        float v = acc[i][j][r] + bv;
        if (EPI == 0) {
          Cf[idx] = v;
        } else if (EPI == 1) {
          Cf[idx] = v + resid[idx];
        } else {
          Ch[idx] = (_Float16)(0.5f * v * (1.f + erff(v * 0.70710678f)));
        }
      }
    }
  }
}

// ---------------- RoPE + layout shuffle ----------------
// qkv f32 [B,S,H,3,64] -> Qh,Kh f16 [B,H,S,64] (rotated),
// Vt f16 [B,H,64,S] (transposed for PV WMMA B-fragments)
__global__ void rope_kernel(const float* __restrict__ qkv,
                            const float* __restrict__ cosT,
                            const float* __restrict__ sinT,
                            _Float16* __restrict__ Qh, _Float16* __restrict__ Kh,
                            _Float16* __restrict__ Vt) {
  int idx = blockIdx.x * blockDim.x + threadIdx.x;  // B*S*H*32
  if (idx >= BATCH * SEQ * NH * 32) return;
  const int pair = idx & 31;
  const int h    = (idx >> 5) & (NH - 1);
  const int s    = (idx >> 9) & (SEQ - 1);
  const int b    = idx >> 20;
  const int d    = pair * 2;
  const size_t base = (((size_t)(b * SEQ + s) * NH + h) * 3) * HD;
  const float q0 = qkv[base + d],        q1 = qkv[base + d + 1];
  const float k0 = qkv[base + HD + d],   k1 = qkv[base + HD + d + 1];
  const float v0 = qkv[base + 2*HD + d], v1 = qkv[base + 2*HD + d + 1];
  const float c  = cosT[s * HD + d];
  const float sn = sinT[s * HD + d];
  const size_t bh = (size_t)(b * NH + h);
  const size_t qo = (bh * SEQ + s) * HD + d;
  Qh[qo]     = (_Float16)(c * q0 - sn * q1);
  Qh[qo + 1] = (_Float16)(c * q1 + sn * q0);
  Kh[qo]     = (_Float16)(c * k0 - sn * k1);
  Kh[qo + 1] = (_Float16)(c * k1 + sn * k0);
  const size_t vo = bh * HD * SEQ;
  Vt[vo + (size_t)d * SEQ + s]       = (_Float16)v0;
  Vt[vo + (size_t)(d + 1) * SEQ + s] = (_Float16)v1;
}

// ---------------- causal flash attention ----------------
// grid: (S/128, B*H), 256 threads (8 waves); wave w owns rows qbase+16w..+15.
// Each staged K/V tile now feeds 128 query rows (half the staging traffic);
// waves skip fully-masked diagonal tiles (wave-uniform scalar branch) and
// only diagonal-straddling tiles pay the cndmask cost.
#define TQ 128
#define TK 32
#define KSTR 72   // Ks row stride in halves (64+8, 16B aligned)
#define VSTR 40
#define PSTR 40

__global__ __launch_bounds__(256)
void attn_kernel(const _Float16* __restrict__ Q, const _Float16* __restrict__ Kh,
                 const _Float16* __restrict__ Vt, _Float16* __restrict__ O) {
  __shared__ __align__(16) _Float16 Ks[TK * KSTR];       // K tile [t][d]
  __shared__ __align__(16) _Float16 Vts[HD * VSTR];      // V^T tile [d][t]
  __shared__ __align__(16) _Float16 Ps[8 * 16 * PSTR];   // per-wave P [16][32]
  const int tid   = threadIdx.x;
  const int lane  = tid & 31;
  const int wave  = tid >> 5;           // 0..7
  const int lr    = lane & 15;
  const int khalf = (lane >> 4) << 3;   // also the +8 row offset in C layout
  const int bh    = blockIdx.y;
  const int qbase = blockIdx.x * TQ;
  const size_t qoff = (size_t)bh * SEQ * HD;

  // staging geometry: exactly one async b128 chunk per thread per tile
  const int kt_t = tid >> 3;            // K tile row 0..31
  const int kt_d = (tid & 7) << 3;      // K tile col (halves)
  const int vt_d = tid >> 2;            // V^T tile row 0..63
  const int vt_t = (tid & 3) << 3;      // V^T tile col (halves)

  // Q fragments (held in registers all loop), pre-scaled by 1/sqrt(hd)
  v16h qf[2];
  {
    const _Float16* qp = Q + qoff + (size_t)(qbase + wave * 16 + lr) * HD;
#pragma unroll
    for (int c = 0; c < 2; c++) {
      v16h f = frag_ld(qp + c * 32, khalf);
#pragma unroll
      for (int e = 0; e < 16; e++) f[e] = f[e] * (_Float16)0.125f;
      qf[c] = f;
    }
  }

  v8f oacc[4];
#pragma unroll
  for (int i = 0; i < 4; i++) oacc[i] = (v8f)0.f;
  float mrow[8], lrow[8];
#pragma unroll
  for (int r = 0; r < 8; r++) { mrow[r] = -__builtin_inff(); lrow[r] = 0.f; }

  const int rowg0   = qbase + wave * 16 + khalf;  // + r = this lane-half's rows
  const int row_min = qbase + wave * 16;          // wave-uniform
  const int row_max = row_min + 15;               // wave-uniform
  const int nkt = qbase / TK + TQ / TK;  // causal: tiles covering t <= qbase+TQ-1
  for (int kt = 0; kt < nkt; kt++) {
    const int kb = kt * TK;
    __syncthreads();
    // async-stage K tile 32x64 and V^T tile 64x32 (ASYNCcnt path)
    ASYNC_LD_B128(lds_off(&Ks[kt_t * KSTR + kt_d]),
                  Kh + qoff + (size_t)(kb + kt_t) * HD + kt_d);
    ASYNC_LD_B128(lds_off(&Vts[vt_d * VSTR + vt_t]),
                  Vt + (size_t)bh * HD * SEQ + (size_t)vt_d * SEQ + kb + vt_t);
    WAIT_ASYNC();
    __syncthreads();

    if (kb > row_max) continue;  // tile entirely above diagonal for this wave

    // scores S = (Q/8) @ K^T : Bt[n=t][k=d] == Ks row-major
    v8f sacc[2];
#pragma unroll
    for (int nt = 0; nt < 2; nt++) {
      v8f s = (v8f)0.f;
#pragma unroll
      for (int c = 0; c < 2; c++) {
        v16h bf = frag_ld(&Ks[(nt * 16 + lr) * KSTR + c * 32], khalf);
        s = __builtin_amdgcn_wmma_f32_16x16x32_f16(false, qf[c], false, bf,
                                                   (short)0, s, false, false);
      }
      sacc[nt] = s;
    }

    // causal mask only on diagonal-straddling tiles (wave-uniform branch)
    if (kb + TK - 1 > row_min) {
#pragma unroll
      for (int r = 0; r < 8; r++) {
        const int rg = rowg0 + r;
#pragma unroll
        for (int nt = 0; nt < 2; nt++) {
          const int cg = kb + nt * 16 + lr;
          if (cg > rg) sacc[nt][r] = -__builtin_inff();
        }
      }
    }

    // online softmax (rows live in 16-lane DPP rows)
    float mnew[8], alpha[8];
#pragma unroll
    for (int r = 0; r < 8; r++) {
      float mx = fmaxf(sacc[0][r], sacc[1][r]);
      mx = red16_max(mx);
      mnew[r]  = fmaxf(mrow[r], mx);
      alpha[r] = __expf(mrow[r] - mnew[r]);
      mrow[r]  = mnew[r];
    }

    _Float16* pp = Ps + wave * 16 * PSTR;
#pragma unroll
    for (int r = 0; r < 8; r++) {
      float rs = 0.f;
#pragma unroll
      for (int nt = 0; nt < 2; nt++) {
        float p = __expf(sacc[nt][r] - mnew[r]);
        rs += p;
        pp[(khalf + r) * PSTR + nt * 16 + lr] = (_Float16)p;  // C-layout -> row-major
      }
      rs = red16_sum(rs);
      lrow[r] = lrow[r] * alpha[r] + rs;
    }
    asm volatile("s_wait_dscnt 0x0" ::: "memory");  // P stores visible within wave

    // rescale output accumulators
#pragma unroll
    for (int dt = 0; dt < 4; dt++)
#pragma unroll
      for (int r = 0; r < 8; r++) oacc[dt][r] *= alpha[r];

    // O += P @ V : A-frag from Ps, B-frag from Vts (Bt[n=d][k=t])
    v16h pf = frag_ld(pp + lr * PSTR, khalf);
#pragma unroll
    for (int dt = 0; dt < 4; dt++) {
      v16h vf = frag_ld(&Vts[(dt * 16 + lr) * VSTR], khalf);
      oacc[dt] = __builtin_amdgcn_wmma_f32_16x16x32_f16(false, pf, false, vf,
                                                        (short)0, oacc[dt], false, false);
    }
  }

  // normalize and write O as f16 [B,S,H*64]
  const int b = bh >> 4, h = bh & (NH - 1);
#pragma unroll
  for (int dt = 0; dt < 4; dt++) {
    const int dc = h * HD + dt * 16 + lr;
#pragma unroll
    for (int r = 0; r < 8; r++) {
      const int sg = qbase + wave * 16 + khalf + r;
      O[((size_t)b * SEQ + sg) * D_MODEL + dc] = (_Float16)(oacc[dt][r] / lrow[r]);
    }
  }
}

// ---------------- orchestration ----------------
extern "C" void kernel_launch(void* const* d_in, const int* in_sizes, int n_in,
                              void* d_out, int out_size, void* d_ws, size_t ws_size,
                              hipStream_t stream) {
  const float* x     = (const float*)d_in[0];
  const float* cosT  = (const float*)d_in[2];
  const float* sinT  = (const float*)d_in[3];
  const float* gAtt  = (const float*)d_in[6];
  const float* gMlp  = (const float*)d_in[7];
  const float* Wqkv  = (const float*)d_in[8];
  const float* bqkv  = (const float*)d_in[9];
  const float* Wo    = (const float*)d_in[10];
  const float* bo    = (const float*)d_in[11];
  const float* Wup   = (const float*)d_in[12];
  const float* bup   = (const float*)d_in[13];
  const float* Wdown = (const float*)d_in[14];
  const float* bdown = (const float*)d_in[15];
  float* out = (float*)d_out;

  char* ws = (char*)d_ws;
  size_t off = 0;
  auto carve = [&](size_t bytes) -> void* {
    void* p = ws + off;
    off += (bytes + 255) & ~(size_t)255;
    return p;
  };
  _Float16* WqkvH  = (_Float16*)carve((size_t)D_MODEL * 3 * D_MODEL * 2);
  _Float16* WoH    = (_Float16*)carve((size_t)D_MODEL * D_MODEL * 2);
  _Float16* WupH   = (_Float16*)carve((size_t)D_MODEL * 4 * D_MODEL * 2);
  _Float16* WdownH = (_Float16*)carve((size_t)4 * D_MODEL * D_MODEL * 2);
  _Float16* h1H    = (_Float16*)carve((size_t)ROWS * D_MODEL * 2);
  float*    qkvF   = (float*)   carve((size_t)ROWS * 3 * D_MODEL * 4);
  _Float16* Qh     = (_Float16*)carve((size_t)ROWS * D_MODEL * 2);
  _Float16* Kh     = (_Float16*)carve((size_t)ROWS * D_MODEL * 2);
  _Float16* VtH    = (_Float16*)carve((size_t)ROWS * D_MODEL * 2);
  _Float16* Oh     = (_Float16*)carve((size_t)ROWS * D_MODEL * 2);
  float*    attnF  = (float*)   carve((size_t)ROWS * D_MODEL * 4);
  _Float16* h2H    = (_Float16*)carve((size_t)ROWS * D_MODEL * 2);
  _Float16* upH    = (_Float16*)carve((size_t)ROWS * 4 * D_MODEL * 2);
  (void)ws_size; (void)in_sizes; (void)n_in; (void)out_size;

  auto cvt = [&](const float* src, _Float16* dst, int n) {
    f32_to_f16_kernel<<<(n + 255) / 256, 256, 0, stream>>>(src, dst, n);
  };
  cvt(Wqkv,  WqkvH,  D_MODEL * 3 * D_MODEL);
  cvt(Wo,    WoH,    D_MODEL * D_MODEL);
  cvt(Wup,   WupH,   D_MODEL * 4 * D_MODEL);
  cvt(Wdown, WdownH, 4 * D_MODEL * D_MODEL);

  // 1) h = rmsnorm(x)*g_attn
  rmsnorm_kernel<<<ROWS, 256, 0, stream>>>(x, gAtt, h1H);
  // 2) qkv = h @ Wqkv + bqkv  (fp32 out for RoPE)
  gemm_wmma<0><<<dim3(3 * D_MODEL / BN, ROWS / BM), 256, 0, stream>>>(
      h1H, WqkvH, bqkv, nullptr, qkvF, nullptr, ROWS, 3 * D_MODEL, D_MODEL);
  // 3) RoPE + head-major relayout
  rope_kernel<<<(BATCH * SEQ * NH * 32 + 255) / 256, 256, 0, stream>>>(
      qkvF, cosT, sinT, Qh, Kh, VtH);
  // 4) causal flash attention
  attn_kernel<<<dim3(SEQ / TQ, BATCH * NH), 256, 0, stream>>>(Qh, Kh, VtH, Oh);
  // 5) attn_out = o @ Wo + bo + x
  gemm_wmma<1><<<dim3(D_MODEL / BN, ROWS / BM), 256, 0, stream>>>(
      Oh, WoH, bo, x, attnF, nullptr, ROWS, D_MODEL, D_MODEL);
  // 6) h2 = rmsnorm(attn_out)*g_mlp
  rmsnorm_kernel<<<ROWS, 256, 0, stream>>>(attnF, gMlp, h2H);
  // 7) up = gelu(h2 @ Wup + bup)  (f16 out)
  gemm_wmma<2><<<dim3(4 * D_MODEL / BN, ROWS / BM), 256, 0, stream>>>(
      h2H, WupH, bup, nullptr, nullptr, upH, ROWS, 4 * D_MODEL, D_MODEL);
  // 8) out = up @ Wdown + bdown + attn_out
  gemm_wmma<1><<<dim3(D_MODEL / BN, ROWS / BM), 256, 0, stream>>>(
      upH, WdownH, bdown, attnF, out, nullptr, ROWS, D_MODEL, 4 * D_MODEL);
}
